// GATBottleneck_2276332667338
// MI455X (gfx1250) — matrix operand order repaired and checked by reading, not compiled, not run on hardware
//
#include <hip/hip_runtime.h>
#include <hip/hip_bf16.h>

typedef __attribute__((ext_vector_type(16))) __bf16 v16bf;
typedef __attribute__((ext_vector_type(8)))  __bf16 v8bf;
typedef __attribute__((ext_vector_type(8)))  float  v8f;

#define HEADS   4
#define IN_CH   128
#define OUT_CH  128
#define HC      512   // HEADS*OUT_CH

// ---------- helpers ----------
__device__ __forceinline__ __bf16 f2bf(float f) {
    unsigned u = __float_as_uint(f);
    u = (u + 0x7FFFu + ((u >> 16) & 1u)) >> 16;   // round-to-nearest-even
    unsigned short s = (unsigned short)u;
    __bf16 b;
    __builtin_memcpy(&b, &s, 2);
    return b;
}

// float atomic-max via sign-split int/uint trick; works with 0xFFFFFFFF init.
__device__ __forceinline__ void atomicMaxFloat(float* addr, float val) {
    if (val >= 0.0f) atomicMax((int*)addr, __float_as_int(val));
    else             atomicMin((unsigned int*)addr, __float_as_uint(val));
}

// 16 contiguous bf16 from LDS -> v16bf (2x ds_load_b128)
__device__ __forceinline__ v16bf ld16(const __bf16* p) {
    v8bf lo = *(const v8bf*)p;
    v8bf hi = *(const v8bf*)(p + 8);
    return __builtin_shufflevector(lo, hi, 0,1,2,3,4,5,6,7,8,9,10,11,12,13,14,15);
}
// A fragment: two separately addressed 8-element halves (ISA K-half layout)
__device__ __forceinline__ v16bf ld8x2(const __bf16* p0, const __bf16* p1) {
    v8bf lo = *(const v8bf*)p0;
    v8bf hi = *(const v8bf*)p1;
    return __builtin_shufflevector(lo, hi, 0,1,2,3,4,5,6,7,8,9,10,11,12,13,14,15);
}

// CDNA5 async global->LDS copy (16B per lane), tracked by ASYNCcnt.
__device__ __forceinline__ void async_ld_b128(unsigned lds_byte_off, const void* gptr) {
    asm volatile("global_load_async_to_lds_b128 %0, %1, off"
                 :: "v"(lds_byte_off), "v"(gptr) : "memory");
}
__device__ __forceinline__ void wait_async0() {
    asm volatile("s_wait_asynccnt 0" ::: "memory");
}
__device__ __forceinline__ unsigned lds_off(const void* p) {
    return (unsigned)(size_t)p;   // low 32 bits of generic addr = LDS byte offset
}

#define WMMA_BF16(A, B, C) \
    __builtin_amdgcn_wmma_f32_16x16x32_bf16(false, (A), false, (B), (short)0, (C), false, false)

// =====================================================================
// Kernel 0: one-time bf16 weight conversion + transpose into workspace.
//   W1T[n][k] = bf16(lin_w[k][n])   (512 x 128)
//   W2T[n][k] = bf16(proj_w[k][n])  (128 x 512)
// =====================================================================
__global__ __launch_bounds__(256) void k_cvt_w(
    const float* __restrict__ lin_w, const float* __restrict__ proj_w,
    __bf16* __restrict__ W1T, __bf16* __restrict__ W2T)
{
    int t = blockIdx.x * 256 + threadIdx.x;
    if (t >= IN_CH * HC) return;
    { // lin_w: [128][512], t = k*512 + n
        int k = t >> 9, n = t & 511;
        W1T[n * IN_CH + k] = f2bf(lin_w[t]);
    }
    { // proj_w: [512][128], t = k*128 + n
        int k = t >> 7, n = t & 127;
        W2T[n * HC + k] = f2bf(proj_w[t]);
    }
}

// =====================================================================
// Kernel 1: H = X @ lin_w   (bf16 WMMA; B pre-transposed, async-staged)
// block = 256 threads (8 waves); tile = 32(M) x 128(N); grid=(M/32, 4)
// Each wave: 2 M-accumulators sharing one B fragment (3 LDS loads / WMMA).
// =====================================================================
__global__ __launch_bounds__(256) void k_gemm1(
    const float* __restrict__ X, const __bf16* __restrict__ W1T,
    float* __restrict__ Hout, int n_nodes)
{
    __shared__ __bf16 Xs[32][IN_CH + 16];    // row stride 288B (16B-mult)
    __shared__ __bf16 BsT[128][IN_CH + 16];  // [n_local][k], row stride 288B

    const int m0 = blockIdx.x * 32;
    const int n0 = blockIdx.y * 128;
    const int tid = threadIdx.x;
    const bool full = (m0 + 32 <= n_nodes);

    // Issue B staging first (async, overlaps with A conversion below)
    for (int idx = tid; idx < 128 * 16; idx += 256) {     // 16B chunks
        int r = idx >> 4, ch = idx & 15;
        async_ld_b128(lds_off(&BsT[r][ch * 8]),
                      W1T + (size_t)(n0 + r) * IN_CH + ch * 8);
    }
    // Stage A: fp32 -> bf16 (32 x 128)
    if (full) {
        for (int idx = tid; idx < 32 * IN_CH; idx += 256) {
            int r = idx >> 7, c = idx & 127;
            Xs[r][c] = f2bf(X[(size_t)(m0 + r) * IN_CH + c]);
        }
    } else {
        for (int idx = tid; idx < 32 * IN_CH; idx += 256) {
            int r = idx >> 7, c = idx & 127;
            float v = (m0 + r < n_nodes) ? X[(size_t)(m0 + r) * IN_CH + c] : 0.0f;
            Xs[r][c] = f2bf(v);
        }
    }
    wait_async0();
    __syncthreads();

    const int wave = tid >> 5, lane = tid & 31;
    const int nt    = wave * 16;
    const int lrow  = lane & 15;
    const int khalf = (lane >> 4) * 8;    // A K-half (ISA layout)
    const int kb    = (lane >> 4) * 16;   // unused; B also loads with khalf

    v8f acc0 = {0.f,0.f,0.f,0.f,0.f,0.f,0.f,0.f};
    v8f acc1 = {0.f,0.f,0.f,0.f,0.f,0.f,0.f,0.f};

    #pragma unroll
    for (int kk = 0; kk < IN_CH; kk += 32) {
        v16bf a0 = ld8x2(&Xs[lrow][kk + khalf],      &Xs[lrow][kk + 16 + khalf]);
        v16bf a1 = ld8x2(&Xs[16 + lrow][kk + khalf], &Xs[16 + lrow][kk + 16 + khalf]);
        v16bf b  = ld8x2(&BsT[nt + lrow][kk + khalf], &BsT[nt + lrow][kk + 16 + khalf]);
        acc0 = WMMA_BF16(a0, b, acc0);
        acc1 = WMMA_BF16(a1, b, acc1);
    }

    const int srow = (lane >> 4) * 8;     // C/D: lanes16-31 hold M+8
    const int scol = lane & 15;
    if (full) {
        #pragma unroll
        for (int r = 0; r < 8; ++r) {
            Hout[(size_t)(m0 + r + srow) * HC + n0 + nt + scol]      = acc0[r];
            Hout[(size_t)(m0 + 16 + r + srow) * HC + n0 + nt + scol] = acc1[r];
        }
    } else {
        #pragma unroll
        for (int r = 0; r < 8; ++r) {
            int row0 = m0 + r + srow, row1 = row0 + 16;
            if (row0 < n_nodes) Hout[(size_t)row0 * HC + n0 + nt + scol] = acc0[r];
            if (row1 < n_nodes) Hout[(size_t)row1 * HC + n0 + nt + scol] = acc1[r];
        }
    }
}

// =====================================================================
// Kernel 2: attention logits a_src / a_dst.  One wave per (node, head).
// =====================================================================
__global__ __launch_bounds__(256) void k_att(
    const float* __restrict__ H, const float* __restrict__ att_src,
    const float* __restrict__ att_dst,
    float* __restrict__ a_src, float* __restrict__ a_dst, int n_nodes)
{
    const int gw   = (blockIdx.x * 256 + threadIdx.x) >> 5;
    const int lane = threadIdx.x & 31;
    const int n = gw >> 2, head = gw & 3;
    if (n >= n_nodes) return;

    const float4 hv = *(const float4*)(H + (size_t)n * HC + head * OUT_CH + lane * 4);
    const float4 as = *(const float4*)(att_src + head * OUT_CH + lane * 4);
    const float4 ad = *(const float4*)(att_dst + head * OUT_CH + lane * 4);
    float ssum = hv.x * as.x + hv.y * as.y + hv.z * as.z + hv.w * as.w;
    float dsum = hv.x * ad.x + hv.y * ad.y + hv.z * ad.z + hv.w * ad.w;
    #pragma unroll
    for (int off = 16; off; off >>= 1) {
        ssum += __shfl_xor(ssum, off, 32);
        dsum += __shfl_xor(dsum, off, 32);
    }
    if (lane == 0) { a_src[n * 4 + head] = ssum; a_dst[n * 4 + head] = dsum; }
}

// =====================================================================
// Kernel 3: e = leaky_relu(a_src[src]+a_dst[dst]); segment max into m.
// =====================================================================
__global__ __launch_bounds__(256) void k_edge_max(
    const int* __restrict__ src_idx, const int* __restrict__ dst_idx,
    const float* __restrict__ a_src, const float* __restrict__ a_dst,
    float* __restrict__ e_buf, float* __restrict__ m_buf,
    int n_edges, int n_nodes)
{
    const int t = blockIdx.x * 256 + threadIdx.x;
    const int total = (n_edges + n_nodes) * 4;
    if (t >= total) return;
    const int e = t >> 2, head = t & 3;
    int s, d;
    if (e < n_edges) { s = src_idx[e]; d = dst_idx[e]; }
    else             { s = d = e - n_edges; }
    float el = a_src[s * 4 + head] + a_dst[d * 4 + head];
    el = el > 0.f ? el : 0.2f * el;
    e_buf[t] = el;
    atomicMaxFloat(m_buf + d * 4 + head, el);
}

// =====================================================================
// Kernel 4: e = exp(e - m[dst]); segment sum into s.
// =====================================================================
__global__ __launch_bounds__(256) void k_edge_exp(
    const int* __restrict__ dst_idx, const float* __restrict__ m_buf,
    float* __restrict__ e_buf, float* __restrict__ s_buf,
    int n_edges, int n_nodes)
{
    const int t = blockIdx.x * 256 + threadIdx.x;
    const int total = (n_edges + n_nodes) * 4;
    if (t >= total) return;
    const int e = t >> 2, head = t & 3;
    const int d = (e < n_edges) ? dst_idx[e] : (e - n_edges);
    float w = __expf(e_buf[t] - m_buf[d * 4 + head]);
    e_buf[t] = w;
    unsafeAtomicAdd(s_buf + d * 4 + head, w);
}

// =====================================================================
// Kernel 5: scatter  agg[dst] += h[src] * alpha.   One wave per edge.
// float4 gathers; head of channel block (lane+32j)*4 is exactly j.
// H (102MB) is L2-resident on MI455X (192MB L2).
// =====================================================================
__global__ __launch_bounds__(256) void k_scatter(
    const int* __restrict__ src_idx, const int* __restrict__ dst_idx,
    const float* __restrict__ H, const float* __restrict__ e_buf,
    const float* __restrict__ s_buf, float* __restrict__ agg,
    int n_edges, int n_nodes)
{
    const int gw   = (blockIdx.x * 256 + threadIdx.x) >> 5;
    const int lane = threadIdx.x & 31;
    const int total = n_edges + n_nodes;
    if (gw >= total) return;
    int s, d;
    if (gw < n_edges) { s = src_idx[gw]; d = dst_idx[gw]; }
    else              { s = d = gw - n_edges; }

    const float4 ev = *(const float4*)(e_buf + (size_t)gw * 4);
    const float4 sv = *(const float4*)(s_buf + (size_t)d * 4);
    float alpha[HEADS] = { ev.x / (sv.x + 1e-16f), ev.y / (sv.y + 1e-16f),
                           ev.z / (sv.z + 1e-16f), ev.w / (sv.w + 1e-16f) };

    const float4* hp = (const float4*)(H + (size_t)s * HC);
    float*        ap = agg + (size_t)d * HC;
    #pragma unroll
    for (int j = 0; j < 4; ++j) {               // head == j for this block
        float4 v = hp[lane + 32 * j];
        float  al = alpha[j];
        int c = (lane + 32 * j) * 4;
        unsafeAtomicAdd(ap + c + 0, v.x * al);
        unsafeAtomicAdd(ap + c + 1, v.y * al);
        unsafeAtomicAdd(ap + c + 2, v.z * al);
        unsafeAtomicAdd(ap + c + 3, v.w * al);
    }
}

// =====================================================================
// Kernel 6 (fused): out = relu(agg+gat_bias) @ proj_w + proj_b + x ;
//                   y = LayerNorm(out).  32 rows/block, full N=128.
// bf16 WMMA; A and B both staged per 128-wide K chunk.
// =====================================================================
__global__ __launch_bounds__(256) void k_gemm2_ln(
    const float* __restrict__ agg, const float* __restrict__ gat_bias,
    const __bf16* __restrict__ W2T, const float* __restrict__ proj_b,
    const float* __restrict__ X, const float* __restrict__ ln_g,
    const float* __restrict__ ln_b, float* __restrict__ Y, int n_nodes)
{
    __shared__ __bf16 As [32][128 + 16];      // K-chunk of A, row stride 288B
    __shared__ __bf16 BsT[128][128 + 16];     // [n][k_local], row stride 288B
    __shared__ float  Out[32][OUT_CH + 4];

    const int m0  = blockIdx.x * 32;
    const int tid = threadIdx.x;
    const bool full = (m0 + 32 <= n_nodes);

    const int wave = tid >> 5, lane = tid & 31;
    const int nt = wave * 16;
    const int lrow = lane & 15, khalf = (lane >> 4) * 8, kb = (lane >> 4) * 16;

    v8f acc0 = {0.f,0.f,0.f,0.f,0.f,0.f,0.f,0.f};
    v8f acc1 = {0.f,0.f,0.f,0.f,0.f,0.f,0.f,0.f};

    for (int k0 = 0; k0 < HC; k0 += 128) {
        __syncthreads();                       // previous chunk consumed
        // B chunk async: BsT[n][0..128) <- W2T[n][k0..k0+128)
        for (int idx = tid; idx < 128 * 16; idx += 256) {   // 16B chunks
            int r = idx >> 4, ch = idx & 15;
            async_ld_b128(lds_off(&BsT[r][ch * 8]),
                          W2T + (size_t)r * HC + k0 + ch * 8);
        }
        // A chunk: relu(agg + gat_bias) -> bf16   (32 x 128)
        if (full) {
            for (int idx = tid; idx < 32 * 128; idx += 256) {
                int r = idx >> 7, c = idx & 127;
                float v = agg[(size_t)(m0 + r) * HC + k0 + c] + gat_bias[k0 + c];
                As[r][c] = f2bf(v > 0.f ? v : 0.f);
            }
        } else {
            for (int idx = tid; idx < 32 * 128; idx += 256) {
                int r = idx >> 7, c = idx & 127;
                float v = 0.f;
                if (m0 + r < n_nodes) {
                    v = agg[(size_t)(m0 + r) * HC + k0 + c] + gat_bias[k0 + c];
                    v = v > 0.f ? v : 0.f;
                }
                As[r][c] = f2bf(v);
            }
        }
        if (k0 + 128 < HC)   // warm L2 for the next chunk (global_prefetch_b8)
            __builtin_prefetch(W2T + (size_t)(tid & 127) * HC + k0 + 128, 0, 1);
        wait_async0();
        __syncthreads();

        #pragma unroll
        for (int kk = 0; kk < 128; kk += 32) {
            v16bf a0 = ld8x2(&As[lrow][kk + khalf],      &As[lrow][kk + 16 + khalf]);
            v16bf a1 = ld8x2(&As[16 + lrow][kk + khalf], &As[16 + lrow][kk + 16 + khalf]);
            v16bf b  = ld8x2(&BsT[nt + lrow][kk + khalf], &BsT[nt + lrow][kk + 16 + khalf]);
            acc0 = WMMA_BF16(a0, b, acc0);
            acc1 = WMMA_BF16(a1, b, acc1);
        }
    }

    // Epilogue: + proj_b + x residual into LDS tile
    {
        const int srow = (lane >> 4) * 8, scol = lane & 15;
        const int col = nt + scol;
        if (full) {
            #pragma unroll
            for (int r = 0; r < 8; ++r) {
                int r0 = r + srow, r1 = r0 + 16;
                Out[r0][col] = acc0[r] + proj_b[col] + X[(size_t)(m0 + r0) * OUT_CH + col];
                Out[r1][col] = acc1[r] + proj_b[col] + X[(size_t)(m0 + r1) * OUT_CH + col];
            }
        } else {
            #pragma unroll
            for (int r = 0; r < 8; ++r) {
                int r0 = r + srow, r1 = r0 + 16;
                float x0 = (m0 + r0 < n_nodes) ? X[(size_t)(m0 + r0) * OUT_CH + col] : 0.f;
                float x1 = (m0 + r1 < n_nodes) ? X[(size_t)(m0 + r1) * OUT_CH + col] : 0.f;
                Out[r0][col] = acc0[r] + proj_b[col] + x0;
                Out[r1][col] = acc1[r] + proj_b[col] + x1;
            }
        }
    }
    __syncthreads();

    // LayerNorm: wave w handles rows 4w .. 4w+3 via shuffle reduction
    #pragma unroll
    for (int rr = 0; rr < 4; ++rr) {
        int row = wave * 4 + rr;
        float sum = 0.f, sq = 0.f;
        #pragma unroll
        for (int j = 0; j < 4; ++j) {
            float v = Out[row][lane + 32 * j];
            sum += v; sq += v * v;
        }
        #pragma unroll
        for (int off = 16; off; off >>= 1) {
            sum += __shfl_xor(sum, off, 32);
            sq  += __shfl_xor(sq,  off, 32);
        }
        float mean = sum * (1.f / OUT_CH);
        float var  = sq  * (1.f / OUT_CH) - mean * mean;
        float rs   = rsqrtf(var + 1e-5f);
        if (m0 + row < n_nodes) {
            #pragma unroll
            for (int j = 0; j < 4; ++j) {
                int c = lane + 32 * j;
                float v = (Out[row][c] - mean) * rs;
                Y[(size_t)(m0 + row) * OUT_CH + c] = v * ln_g[c] + ln_b[c];
            }
        }
    }
}

// =====================================================================
extern "C" void kernel_launch(void* const* d_in, const int* in_sizes, int n_in,
                              void* d_out, int out_size, void* d_ws, size_t ws_size,
                              hipStream_t stream) {
    const float* x        = (const float*)d_in[0];
    const int*   edge     = (const int*)  d_in[1];   // [2, E]
    const float* lin_w    = (const float*)d_in[2];
    const float* att_src  = (const float*)d_in[3];
    const float* att_dst  = (const float*)d_in[4];
    const float* gat_bias = (const float*)d_in[5];
    const float* proj_w   = (const float*)d_in[6];
    const float* proj_b   = (const float*)d_in[7];
    const float* ln_g     = (const float*)d_in[8];
    const float* ln_b     = (const float*)d_in[9];
    float* Y = (float*)d_out;

    const int n_nodes = in_sizes[0] / IN_CH;
    const int n_edges = in_sizes[1] / 2;
    const int etot    = n_edges + n_nodes;          // with self-loops

    const int* src_idx = edge;
    const int* dst_idx = edge + n_edges;

    // ---- workspace layout ----
    char* ws = (char*)d_ws;
    size_t off = 0;
    auto take = [&](size_t bytes) -> char* {
        char* p = ws + off;
        off = (off + bytes + 255) & ~(size_t)255;
        return p;
    };
    float*  Hbuf = (float*)take((size_t)n_nodes * HC * sizeof(float));   // 102.4 MB
    float*  aggb = (float*)take((size_t)n_nodes * HC * sizeof(float));   // 102.4 MB
    float*  asrc = (float*)take((size_t)n_nodes * 4 * sizeof(float));
    float*  adst = (float*)take((size_t)n_nodes * 4 * sizeof(float));
    float*  mbuf = (float*)take((size_t)n_nodes * 4 * sizeof(float));
    float*  sbuf = (float*)take((size_t)n_nodes * 4 * sizeof(float));
    float*  ebuf = (float*)take((size_t)etot * 4 * sizeof(float));       // 13.6 MB
    __bf16* W1T  = (__bf16*)take((size_t)HC * IN_CH * sizeof(__bf16));   // 128 KB
    __bf16* W2T  = (__bf16*)take((size_t)OUT_CH * HC * sizeof(__bf16));  // 128 KB

    hipMemsetAsync(aggb, 0,    (size_t)n_nodes * HC * sizeof(float), stream);
    hipMemsetAsync(sbuf, 0,    (size_t)n_nodes * 4  * sizeof(float), stream);
    hipMemsetAsync(mbuf, 0xFF, (size_t)n_nodes * 4  * sizeof(float), stream); // -inf sentinel

    // 0) bf16 weight conversion + transpose
    k_cvt_w<<<(IN_CH * HC + 255) / 256, 256, 0, stream>>>(lin_w, proj_w, W1T, W2T);

    // 1) H = X @ lin_w
    dim3 g1((n_nodes + 31) / 32, HC / 128);
    k_gemm1<<<g1, 256, 0, stream>>>(x, W1T, Hbuf, n_nodes);

    // 2) attention logits
    int att_waves = n_nodes * HEADS;
    k_att<<<(att_waves + 7) / 8, 256, 0, stream>>>(Hbuf, att_src, att_dst,
                                                   asrc, adst, n_nodes);

    // 3) edge logits + segment max
    int et4 = etot * 4;
    k_edge_max<<<(et4 + 255) / 256, 256, 0, stream>>>(src_idx, dst_idx, asrc, adst,
                                                      ebuf, mbuf, n_edges, n_nodes);

    // 4) exp + segment sum
    k_edge_exp<<<(et4 + 255) / 256, 256, 0, stream>>>(dst_idx, mbuf, ebuf, sbuf,
                                                      n_edges, n_nodes);

    // 5) weighted scatter-aggregate (one wave per edge)
    k_scatter<<<(etot + 7) / 8, 256, 0, stream>>>(src_idx, dst_idx, Hbuf, ebuf,
                                                  sbuf, aggb, n_edges, n_nodes);

    // 6) fused projection GEMM + residual + LayerNorm
    k_gemm2_ln<<<(n_nodes + 31) / 32, 256, 0, stream>>>(aggb, gat_bias, W2T, proj_b,
                                                        x, ln_g, ln_b, Y, n_nodes);
}
